// InpaintingAttnProcessor2_0_24661702214004
// MI455X (gfx1250) — matrix operand order repaired and checked by reading, not compiled, not run on hardware
//
#include <hip/hip_runtime.h>
#include <hip/hip_bf16.h>

typedef __bf16 bfloat;
typedef bfloat bf16x8  __attribute__((ext_vector_type(8)));
typedef bfloat bf16x16 __attribute__((ext_vector_type(16)));
typedef float  f32x8   __attribute__((ext_vector_type(8)));

#define S_LEN 4096
#define C_DIM 640
#define NEGBIG (-1e9f)

__device__ __forceinline__ bf16x8 ld8(const bfloat* p) {
  return *(const bf16x8*)p;
}
__device__ __forceinline__ bf16x8 zero8() {
  bf16x8 z;
#pragma unroll
  for (int i = 0; i < 8; ++i) z[i] = (bfloat)0.0f;
  return z;
}
__device__ __forceinline__ bf16x16 cat16(bf16x8 lo, bf16x8 hi) {
  return __builtin_shufflevector(lo, hi, 0,1,2,3,4,5,6,7,8,9,10,11,12,13,14,15);
}
__device__ __forceinline__ f32x8 wmma_bf16(bf16x16 a, bf16x16 b, f32x8 c) {
  // D = A(16x32 bf16) * B(32x16 bf16) + C(16x16 f32)
  return __builtin_amdgcn_wmma_f32_16x16x32_bf16(false, a, false, b, (short)0, c, false, false);
}
__device__ __forceinline__ f32x8 zacc() {
  f32x8 a;
#pragma unroll
  for (int i = 0; i < 8; ++i) a[i] = 0.0f;
  return a;
}

// ---------------- mask prep: nearest resize 512x512 -> 64x64 ----------------
__global__ void k_masks(const int* __restrict__ mask, const int* __restrict__ inp,
                        int* __restrict__ mlab, int* __restrict__ im0) {
  int i = blockIdx.x * blockDim.x + threadIdx.x;
  if (i >= S_LEN) return;
  int r = i >> 6, c = i & 63;
  int src = (r * 8) * 512 + c * 8;   // floor(i*512/64) = i*8
  mlab[i] = mask[src];
  im0[i]  = (inp[src] == 0) ? 1 : 0;
}

// ---------------- f32 -> bf16 elementwise cast ----------------
__global__ void k_cast(const float* __restrict__ src, bfloat* __restrict__ dst, int n) {
  int i = blockIdx.x * blockDim.x + threadIdx.x;
  if (i < n) dst[i] = (bfloat)src[i];
}

struct W13 { const float* w[13]; };
__global__ void k_castw(W13 wp, bfloat* __restrict__ dst) {
  int i = blockIdx.x * blockDim.x + threadIdx.x;
  int z = blockIdx.y;
  if (i < C_DIM * C_DIM)
    dst[(size_t)z * (C_DIM * C_DIM) + i] = (bfloat)wp.w[z][i];
}

// ---------------- 9 QKV projections: out = X @ W^T (32x32 tile / wave) --------
// z%3==2 (the V projections) are written transposed [C, S] for P*V B-operands.
struct P9 { bfloat* p[9]; };
__global__ __launch_bounds__(32)
void k_proj(const bfloat* __restrict__ X, const bfloat* __restrict__ Wall, P9 outs) {
  const int lane = threadIdx.x & 31;
  const int li = lane & 15, hi = lane >> 4;
  const int m0 = blockIdx.x * 32, n0 = blockIdx.y * 32;
  const int z  = blockIdx.z;
  const bfloat* W = Wall + (size_t)z * (C_DIM * C_DIM);

  f32x8 acc00 = zacc(), acc01 = zacc(), acc10 = zacc(), acc11 = zacc();

  const bfloat* a0p = X + (size_t)(m0 + li) * C_DIM;        // A: lane = row m
  const bfloat* a1p = X + (size_t)(m0 + 16 + li) * C_DIM;
  const bfloat* b0p = W + (size_t)(n0 + li) * C_DIM;        // B: lane = col n
  const bfloat* b1p = W + (size_t)(n0 + 16 + li) * C_DIM;
#pragma unroll 5
  for (int s = 0; s < C_DIM; s += 32) {
    bf16x16 a0 = cat16(ld8(a0p + s + 8 * hi), ld8(a0p + s + 16 + 8 * hi));
    bf16x16 a1 = cat16(ld8(a1p + s + 8 * hi), ld8(a1p + s + 16 + 8 * hi));
    bf16x16 b0 = cat16(ld8(b0p + s + 8 * hi), ld8(b0p + s + 16 + 8 * hi));
    bf16x16 b1 = cat16(ld8(b1p + s + 8 * hi), ld8(b1p + s + 16 + 8 * hi));
    acc00 = wmma_bf16(a0, b0, acc00);
    acc01 = wmma_bf16(a0, b1, acc01);
    acc10 = wmma_bf16(a1, b0, acc10);
    acc11 = wmma_bf16(a1, b1, acc11);
  }

  bfloat* dst = outs.p[z];
  const bool tr = (z % 3) == 2;
  auto emit = [&](const f32x8& acc, int mb, int nb) {
#pragma unroll
    for (int r = 0; r < 8; ++r) {
      int M = mb + r + 8 * hi;    // C/D layout: row = r + 8*hi
      int N = nb + li;
      bfloat v = (bfloat)acc[r];
      if (tr) dst[(size_t)N * S_LEN + M] = v;    // V^T [C, S]
      else    dst[(size_t)M * C_DIM + N] = v;    // row-major [S, C]
    }
  };
  emit(acc00, m0, n0);
  emit(acc01, m0, n0 + 16);
  emit(acc10, m0 + 16, n0);
  emit(acc11, m0 + 16, n0 + 16);
}

// ---------------- generic masked flash-attention ----------------
// One wave per 16-query tile; blockIdx.y = head (KD=80 branches) or NT*16-wide
// output slice (entity branch, KD=640 with redundant QK recompute per slice).
template <int KD, int NT, int OUTSIDE>
__global__ __launch_bounds__(32)
void k_attn(const bfloat* __restrict__ Q, const bfloat* __restrict__ K,
            const bfloat* __restrict__ VT, const int* __restrict__ mlab,
            const int* __restrict__ im0, float* __restrict__ out,
            int qk_stride, float scale, int accum) {
  constexpr int NFULL = KD / 32;               // full K=32 steps
  constexpr bool TAIL = (KD % 32) != 0;        // 16-wide tail (KD=80)
  constexpr int NKS = NFULL + (TAIL ? 1 : 0);
  constexpr bool HOIST_Q = (NKS <= 3);
  __shared__ float plds[16 * 32];

  const int lane = threadIdx.x & 31;
  const int li = lane & 15, hi = lane >> 4;
  const int q0 = blockIdx.x * 16;
  const int qkoff = (int)blockIdx.y * qk_stride;  // column offset into Q/K
  const int voff  = (int)blockIdx.y * (NT * 16);  // output-column / V^T-row offset

  int qlab[8];
#pragma unroll
  for (int r = 0; r < 8; ++r) qlab[r] = mlab[q0 + r + 8 * hi];

  float mstate[8], lstate[8];
  f32x8 oacc[NT];
#pragma unroll
  for (int r = 0; r < 8; ++r) { mstate[r] = -3.0e38f; lstate[r] = 0.0f; }
#pragma unroll
  for (int nt = 0; nt < NT; ++nt) oacc[nt] = zacc();

  const bfloat* qrow = Q + (size_t)(q0 + li) * C_DIM + qkoff;

  // hoist Q fragments for small KD (A-chunk validity is uniform per step)
  bf16x16 qf[HOIST_Q ? NKS : 1];
  if constexpr (HOIST_Q) {
#pragma unroll
    for (int s = 0; s < NFULL; ++s)
      qf[s] = cat16(ld8(qrow + s * 32 + 8 * hi), ld8(qrow + s * 32 + 16 + 8 * hi));
    if constexpr (TAIL)
      qf[NFULL] = cat16(ld8(qrow + NFULL * 32 + 8 * hi), zero8());
  }

  for (int j0 = 0; j0 < S_LEN; j0 += 32) {
    const bfloat* k0row = K + (size_t)(j0 + li) * C_DIM + qkoff;
    const bfloat* k1row = K + (size_t)(j0 + 16 + li) * C_DIM + qkoff;

    if (j0 + 32 < S_LEN)   // prefetch next key tile's rows (global_prefetch_b8)
      __builtin_prefetch((const void*)(K + (size_t)(j0 + 32 + lane) * C_DIM + qkoff), 0, 0);

    // key-tile metadata (lane = key column within subtile)
    int lab0 = mlab[j0 + li];
    int lab1 = mlab[j0 + 16 + li];
    bool oks0 = true, oks1 = true;
    if constexpr (OUTSIDE) {
      oks0 = (im0[j0 + li] != 0);
      oks1 = (im0[j0 + 16 + li] != 0);
    }

    // ---- scores S = Q K^T ----
    f32x8 s0 = zacc(), s1 = zacc();
#pragma unroll 4
    for (int s = 0; s < NFULL; ++s) {
      bf16x16 a;
      if constexpr (HOIST_Q) a = qf[s];
      else a = cat16(ld8(qrow + s * 32 + 8 * hi), ld8(qrow + s * 32 + 16 + 8 * hi));
      int bb = s * 32 + 8 * hi;
      s0 = wmma_bf16(a, cat16(ld8(k0row + bb), ld8(k0row + bb + 16)), s0);
      s1 = wmma_bf16(a, cat16(ld8(k1row + bb), ld8(k1row + bb + 16)), s1);
    }
    if constexpr (TAIL) {   // last 16 of K-dim: elements 8..15 are out of range
      bf16x16 a;
      if constexpr (HOIST_Q) a = qf[NFULL];
      else a = cat16(ld8(qrow + NFULL * 32 + 8 * hi), zero8());
      int bb = NFULL * 32 + 8 * hi;
      bf16x8 b0l = ld8(k0row + bb);
      bf16x8 b0h = zero8();
      s0 = wmma_bf16(a, cat16(b0l, b0h), s0);
      bf16x8 b1l = ld8(k1row + bb);
      bf16x8 b1h = zero8();
      s1 = wmma_bf16(a, cat16(b1l, b1h), s1);
    }

    // ---- mask + online softmax (per row r+8*hi, reduce across 16-lane half) ----
#pragma unroll
    for (int r = 0; r < 8; ++r) {
      float v0 = (qlab[r] == lab0 && oks0) ? s0[r] * scale : NEGBIG;
      float v1 = (qlab[r] == lab1 && oks1) ? s1[r] * scale : NEGBIG;
      float rm = fmaxf(v0, v1);
      rm = fmaxf(rm, __shfl_xor(rm, 1, 32));
      rm = fmaxf(rm, __shfl_xor(rm, 2, 32));
      rm = fmaxf(rm, __shfl_xor(rm, 4, 32));
      rm = fmaxf(rm, __shfl_xor(rm, 8, 32));
      float mnew  = fmaxf(mstate[r], rm);
      float alpha = __expf(mstate[r] - mnew);
      float p0 = __expf(v0 - mnew);
      float p1 = __expf(v1 - mnew);
      float rs = p0 + p1;
      rs += __shfl_xor(rs, 1, 32);
      rs += __shfl_xor(rs, 2, 32);
      rs += __shfl_xor(rs, 4, 32);
      rs += __shfl_xor(rs, 8, 32);
      lstate[r] = lstate[r] * alpha + rs;
      mstate[r] = mnew;
#pragma unroll
      for (int nt = 0; nt < NT; ++nt) oacc[nt][r] *= alpha;
      int m = r + 8 * hi;
      plds[m * 32 + li]      = p0;
      plds[m * 32 + 16 + li] = p1;
    }
    __syncthreads();

    // ---- rebuild P as a 16x32 bf16 A-fragment from LDS ----
    bf16x16 pfrag;
#pragma unroll
    for (int r = 0; r < 8; ++r) {
      int kb = 16 * (r >> 2) + 8 * hi + 2 * (r & 3);   // A-layout K pair
      float2 pv = *(const float2*)&plds[li * 32 + kb];
      pfrag[2 * r]     = (bfloat)pv.x;
      pfrag[2 * r + 1] = (bfloat)pv.y;
    }

    // ---- O += P * V  (B from V^T: lane = d column, K=j contiguous) ----
#pragma unroll
    for (int nt = 0; nt < NT; ++nt) {
      const bfloat* vp = VT + (size_t)(voff + nt * 16 + li) * S_LEN + j0 + 8 * hi;
      oacc[nt] = wmma_bf16(pfrag, cat16(ld8(vp), ld8(vp + 16)), oacc[nt]);
    }
    __syncthreads();
  }

  // ---- normalize and emit ----
#pragma unroll
  for (int r = 0; r < 8; ++r) {
    float inv = 1.0f / lstate[r];
    int m = q0 + r + 8 * hi;
#pragma unroll
    for (int nt = 0; nt < NT; ++nt) {
      size_t idx = (size_t)m * C_DIM + voff + nt * 16 + li;
      float val = oacc[nt][r] * inv;
      if (accum) out[idx] += val;   // kernels serialize on stream; disjoint in-kernel
      else       out[idx] = val;
    }
  }
}

// ---------------- output projection: out = HS @ Wo^T + residual (32x32/wave) ---
__global__ __launch_bounds__(32)
void k_oproj(const bfloat* __restrict__ Xb, const bfloat* __restrict__ Wo,
             const float* __restrict__ resid, float* __restrict__ out) {
  const int lane = threadIdx.x & 31;
  const int li = lane & 15, hi = lane >> 4;
  const int m0 = blockIdx.x * 32, n0 = blockIdx.y * 32;

  f32x8 acc00 = zacc(), acc01 = zacc(), acc10 = zacc(), acc11 = zacc();

  const bfloat* a0p = Xb + (size_t)(m0 + li) * C_DIM;
  const bfloat* a1p = Xb + (size_t)(m0 + 16 + li) * C_DIM;
  const bfloat* b0p = Wo + (size_t)(n0 + li) * C_DIM;
  const bfloat* b1p = Wo + (size_t)(n0 + 16 + li) * C_DIM;
#pragma unroll 5
  for (int s = 0; s < C_DIM; s += 32) {
    bf16x16 a0 = cat16(ld8(a0p + s + 8 * hi), ld8(a0p + s + 16 + 8 * hi));
    bf16x16 a1 = cat16(ld8(a1p + s + 8 * hi), ld8(a1p + s + 16 + 8 * hi));
    bf16x16 b0 = cat16(ld8(b0p + s + 8 * hi), ld8(b0p + s + 16 + 8 * hi));
    bf16x16 b1 = cat16(ld8(b1p + s + 8 * hi), ld8(b1p + s + 16 + 8 * hi));
    acc00 = wmma_bf16(a0, b0, acc00);
    acc01 = wmma_bf16(a0, b1, acc01);
    acc10 = wmma_bf16(a1, b0, acc10);
    acc11 = wmma_bf16(a1, b1, acc11);
  }
  auto emit = [&](const f32x8& acc, int mb, int nb) {
#pragma unroll
    for (int r = 0; r < 8; ++r) {
      size_t idx = (size_t)(mb + r + 8 * hi) * C_DIM + nb + li;
      out[idx] = acc[r] + resid[idx];
    }
  };
  emit(acc00, m0, n0);
  emit(acc01, m0, n0 + 16);
  emit(acc10, m0 + 16, n0);
  emit(acc11, m0 + 16, n0 + 16);
}

extern "C" void kernel_launch(void* const* d_in, const int* in_sizes, int n_in,
                              void* d_out, int out_size, void* d_ws, size_t ws_size,
                              hipStream_t stream) {
  (void)in_sizes; (void)n_in; (void)out_size; (void)ws_size;
  const float* hidden = (const float*)d_in[0];
  const int*   mask   = (const int*)d_in[1];
  const int*   inp    = (const int*)d_in[2];

  char* ws = (char*)d_ws;
  size_t off = 0;
  auto take = [&](size_t bytes) -> void* {
    void* p = ws + off;
    off += (bytes + 255) & ~(size_t)255;
    return p;
  };
  const size_t NE = (size_t)S_LEN * C_DIM;

  bfloat* Xb   = (bfloat*)take(NE * 2);
  bfloat* qb   = (bfloat*)take(NE * 2);
  bfloat* kb   = (bfloat*)take(NE * 2);
  bfloat* vtb  = (bfloat*)take(NE * 2);
  bfloat* qeb  = (bfloat*)take(NE * 2);
  bfloat* keb  = (bfloat*)take(NE * 2);
  bfloat* vetb = (bfloat*)take(NE * 2);
  bfloat* qob  = (bfloat*)take(NE * 2);
  bfloat* kob  = (bfloat*)take(NE * 2);
  bfloat* votb = (bfloat*)take(NE * 2);
  bfloat* Wb   = (bfloat*)take((size_t)13 * C_DIM * C_DIM * 2);
  float*  hacc = (float*)take(NE * 4);
  bfloat* hsb  = (bfloat*)take(NE * 2);
  int*    mlab = (int*)take(S_LEN * 4);
  int*    im0  = (int*)take(S_LEN * 4);

  k_masks<<<(S_LEN + 255) / 256, 256, 0, stream>>>(mask, inp, mlab, im0);
  k_cast<<<(int)((NE + 255) / 256), 256, 0, stream>>>(hidden, Xb, (int)NE);

  W13 wp;
  for (int i = 0; i < 13; ++i) wp.w[i] = (const float*)d_in[3 + i];
  k_castw<<<dim3((C_DIM * C_DIM + 255) / 256, 13), 256, 0, stream>>>(wp, Wb);

  P9 outs = {{qb, kb, vtb, qeb, keb, vetb, qob, kob, votb}};
  k_proj<<<dim3(S_LEN / 32, C_DIM / 32, 9), 32, 0, stream>>>(Xb, Wb, outs);

  const float sc8 = 0.11180339887498948f;   // 1/sqrt(80)
  const float sc1 = 0.03952847075210474f;   // 1/sqrt(640)
  // branch 0: 8-head attention (write)
  k_attn<80, 5, 0><<<dim3(S_LEN / 16, 8), 32, 0, stream>>>(
      qb, kb, vtb, mlab, im0, hacc, 80, sc8, 0);
  // branch 1: entity attention, 1 head d=640 in 4 slices of 160 cols (accumulate)
  k_attn<640, 10, 0><<<dim3(S_LEN / 16, 4), 32, 0, stream>>>(
      qeb, keb, vetb, mlab, im0, hacc, 0, sc1, 1);
  // branch 2: outside attention (accumulate)
  k_attn<80, 5, 1><<<dim3(S_LEN / 16, 8), 32, 0, stream>>>(
      qob, kob, votb, mlab, im0, hacc, 80, sc8, 1);

  k_cast<<<(int)((NE + 255) / 256), 256, 0, stream>>>(hacc, hsb, (int)NE);
  k_oproj<<<dim3(S_LEN / 32, C_DIM / 32), 32, 0, stream>>>(
      hsb, Wb + (size_t)12 * C_DIM * C_DIM, hidden, (float*)d_out);
}
